// SpaTrans_50843822850673
// MI455X (gfx1250) — matrix operand order repaired and checked by reading, not compiled, not run on hardware
//
#include <hip/hip_runtime.h>
#include <hip/hip_bf16.h>

// ---------------------------------------------------------------------------
// SpaTrans forward for MI455X (gfx1250, wave32, WMMA).
// All GEMMs use v_wmma_f32_16x16x32_bf16 (bf16 inputs, f32 accumulate);
// residual / LayerNorm chains stay in f32.  Workspace layout (~400MB):
//   [0,100MB)   T0   f32 [R,256]  spa_token residual
//   [100,200)   TS/Y1 f32 [R,256]
//   [200,250)   X1/AO/Y2b bf16 [R,256]
//   [250,400)   QKV bf16 [R,768]  (reused as X2 [R,256]b + G [R,512]b)
// ---------------------------------------------------------------------------

typedef __attribute__((ext_vector_type(8)))  __bf16 bf16x8;
typedef __attribute__((ext_vector_type(16))) __bf16 bf16x16;
typedef __attribute__((ext_vector_type(8)))  float  f32x8;

union FragAB { bf16x16 v; struct { bf16x8 lo, hi; } p; };

__device__ __forceinline__ f32x8 wmma_bf16f32(bf16x16 a, bf16x16 b, f32x8 c) {
  return __builtin_amdgcn_wmma_f32_16x16x32_bf16(false, a, false, b,
                                                 (short)0, c, false, false);
}
// A-operand (16x32, M rows across lanes%16): per-lane k = (e&7)+8*half+16*(e>>3)
__device__ __forceinline__ bf16x16 ld_frag_a(const __bf16* rowp, int half) {
  FragAB f;
  f.p.lo = *(const bf16x8*)(rowp + 8 * half);
  f.p.hi = *(const bf16x8*)(rowp + 16 + 8 * half);
  return f.v;
}
// B-operand (32x16, N cols across lanes%16): per-lane k = (e&7)+8*half+16*(e>>3)
__device__ __forceinline__ bf16x16 ld_frag_b(const __bf16* rowp, int half) {
  FragAB f;
  f.p.lo = *(const bf16x8*)(rowp + 8 * half);
  f.p.hi = *(const bf16x8*)(rowp + 16 + 8 * half);
  return f.v;
}
__device__ __forceinline__ f32x8 zero8f() {
  f32x8 z; for (int i = 0; i < 8; ++i) z[i] = 0.0f; return z;
}
__device__ __forceinline__ bf16x8 zero8b() {
  bf16x8 z; for (int i = 0; i < 8; ++i) z[i] = (__bf16)0.0f; return z;
}

// ---------------------------------------------------------------------------
// Generic bf16 GEMM:  C[M,N] = A[M,K] * B[K,N] (+bias) (+relu) (+resid)
//   a_mode 0: A from bf16 buffer (lda)
//   a_mode 1: A = im2col(3x3, pad 1) of As0            (tokenization)
//   a_mode 2: A = im2col of (As0 + As1)                (token + PE, linearity)
//   bT: B given as [N,K] row-major (Wconv) instead of [K,N]
//   scatter: store to [b,128,25,32,32] conv output layout
// Block 256 = 8 waves, tile 64M x 128N, wave tile 16M x 64N, K-step 32.
// ---------------------------------------------------------------------------
__global__ __launch_bounds__(256) void gemm_bf16_kernel(
    const __bf16* __restrict__ Abf, const float* __restrict__ As0,
    const float* __restrict__ As1, int a_mode, int lda,
    const float* __restrict__ Bw, int ldb, int bT, int N, int K,
    const float* __restrict__ bias, const float* __restrict__ resid,
    int relu, float* __restrict__ outF, __bf16* __restrict__ outB,
    int scatter)
{
  __shared__ __align__(16) __bf16 Asub[64][40];
  __shared__ __align__(16) __bf16 Bt[128][40];

  const int t = threadIdx.x;
  const int lane = t & 31, wave = t >> 5;
  const int half = lane >> 4, col = lane & 15;
  const int wr = wave >> 1, wc = wave & 1;
  const int m0 = blockIdx.x * 64;
  const int n0 = blockIdx.y * 128;

  f32x8 acc[4];
  for (int f = 0; f < 4; ++f) acc[f] = zero8f();

  for (int k0 = 0; k0 < K; k0 += 32) {
    // ---- stage A tile [64][32] ----
    {
      int row = t >> 2, cb = (t & 3) * 8;
      if (a_mode == 0) {
        bf16x8 v = *(const bf16x8*)(Abf + (size_t)(m0 + row) * lda + k0 + cb);
        *(bf16x8*)&Asub[row][cb] = v;
      } else {
        int r = m0 + row;
        int hwi = r / 100, ba = r - hwi * 100;
        int bb = ba / 25, aa = ba - bb * 25;
        int y0 = hwi >> 5, x0 = hwi & 31;
        for (int e = 0; e < 8; ++e) {
          int k = k0 + cb + e;
          int c = k / 9, rem = k - c * 9;
          int ki = rem / 3, kj = rem - ki * 3;
          int y = y0 + ki - 1, x = x0 + kj - 1;
          float v = 0.0f;
          if (y >= 0 && y < 32 && x >= 0 && x < 32) {
            size_t off = (((size_t)bb * 128 + c) * 25 + aa) * 1024 + y * 32 + x;
            v = As0[off];
            if (a_mode == 2) v += As1[off];
          }
          Asub[row][cb + e] = (__bf16)v;
        }
      }
    }
    // ---- stage B tile -> Bt[n][k] (pre-transposed for B-operand loads) ----
    if (!bT) {
      int k = t >> 3, nb = (t & 7) * 16;
      const float* bp = Bw + (size_t)(k0 + k) * ldb + n0 + nb;
      for (int e = 0; e < 16; ++e) Bt[nb + e][k] = (__bf16)bp[e];
    } else {
      int n = t >> 1, kb = (t & 1) * 16;
      const float* bp = Bw + (size_t)(n0 + n) * ldb + k0 + kb;
      for (int e = 0; e < 16; ++e) Bt[n][kb + e] = (__bf16)bp[e];
    }
    __syncthreads();
    // ---- 4 WMMAs per wave per K-step ----
    bf16x16 a = ld_frag_a(&Asub[wr * 16 + col][0], half);
    for (int f = 0; f < 4; ++f) {
      bf16x16 b = ld_frag_b(&Bt[wc * 64 + f * 16 + col][0], half);
      acc[f] = wmma_bf16f32(a, b, acc[f]);
    }
    __syncthreads();
  }

  // ---- epilogue (C layout: lane%16 = N col, rows = half*8 + vgpr) ----
  for (int f = 0; f < 4; ++f) {
    int nc = n0 + wc * 64 + f * 16 + col;
    for (int r = 0; r < 8; ++r) {
      int mr = m0 + wr * 16 + half * 8 + r;
      float v = acc[f][r];
      if (bias)  v += bias[nc];
      if (relu)  v = v > 0.0f ? v : 0.0f;
      if (resid) v += resid[(size_t)mr * N + nc];
      if (outF) {
        if (scatter) {
          int hwi = mr / 100, ba = mr - hwi * 100;
          int bb = ba / 25, aa = ba - bb * 25;
          outF[(((size_t)bb * 128 + nc) * 25 + aa) * 1024 + hwi] = v;
        } else {
          outF[(size_t)mr * N + nc] = v;
        }
      }
      if (outB) outB[(size_t)mr * N + nc] = (__bf16)v;
    }
  }
}

// ---------------------------------------------------------------------------
// LayerNorm over 256 cols; one wave (32 lanes x 8 elems) per row -> bf16 out.
// ---------------------------------------------------------------------------
__global__ __launch_bounds__(256) void ln_kernel(
    const float* __restrict__ X, const float* __restrict__ g,
    const float* __restrict__ b, __bf16* __restrict__ out)
{
  int t = threadIdx.x, lane = t & 31, wave = t >> 5;
  size_t row = (size_t)blockIdx.x * 8 + wave;
  const float* xp = X + row * 256;
  float v[8], s = 0.f, sq = 0.f;
  for (int i = 0; i < 8; ++i) {
    v[i] = xp[lane + i * 32];
    s += v[i]; sq += v[i] * v[i];
  }
  for (int off = 16; off > 0; off >>= 1) {
    s  += __shfl_xor(s, off, 32);
    sq += __shfl_xor(sq, off, 32);
  }
  float mean = s * (1.0f / 256.0f);
  float var  = sq * (1.0f / 256.0f) - mean * mean;
  float rs   = rsqrtf(var + 1e-5f);
  __bf16* op = out + row * 256;
  for (int i = 0; i < 8; ++i) {
    int c = lane + i * 32;
    op[c] = (__bf16)((v[i] - mean) * rs * g[c] + b[c]);
  }
}

// ---------------------------------------------------------------------------
// Attention: one workgroup per (hw-batch Bq, 16-row n-tile). N=100 padded 112.
// Pre-softmax talking-heads folded into Q:  L[g,n,m] = (s*Wl[g,h]*q_h) . k (K=256).
// bl is softmax-shift-invariant -> dropped. Wave g owns head g.
// Post-softmax mix + attnscale done elementwise via LDS P buffer, then A@V.
// LDS = 61440 (K/V union) + 8448 (Q) + 57344 (P) + 34816 (A) = 158.25 KB.
// ---------------------------------------------------------------------------
__global__ __launch_bounds__(256) void attn_kernel(
    const __bf16* __restrict__ QKV, const float* __restrict__ Wl,
    const float* __restrict__ Ww, const float* __restrict__ bw,
    const float* __restrict__ lamb, __bf16* __restrict__ AO)
{
  __shared__ __align__(16) char kv_u[61440];          // Ksm then (reused) Vt
  __bf16 (*Ksm)[272] = (__bf16(*)[272])kv_u;          // [112][272]  K[m][kk]
  __bf16 (*Vt)[120]  = (__bf16(*)[120])kv_u;          // [256][120]  V^T[col][m]
  __shared__ __align__(16) __bf16 Qsm[16][264];
  __shared__ float  Pbuf[8][16][112];
  __shared__ __align__(16) __bf16 Abuf[8][16][136];

  const int t = threadIdx.x;
  const int lane = t & 31, g = t >> 5;
  const int half = lane >> 4, col = lane & 15;
  const int Bq = blockIdx.x;
  const int n0 = blockIdx.y * 16;
  const __bf16* qkv = QKV + (size_t)Bq * 100 * 768;
  const float scale = 0.17677669529663688f;  // 32^-0.5

  // ---- stage K [112][256] (natural = B-operand staging) and Q [16][256] ----
  for (int idx = t; idx < 112 * 32; idx += 256) {
    int m = idx >> 5, kb = (idx & 31) * 8;
    bf16x8 v = zero8b();
    if (m < 100) v = *(const bf16x8*)(qkv + (size_t)m * 768 + 256 + kb);
    *(bf16x8*)&Ksm[m][kb] = v;
  }
  for (int idx = t; idx < 16 * 32; idx += 256) {
    int n = idx >> 5, kb = (idx & 31) * 8;
    int nn = n0 + n;
    bf16x8 v = zero8b();
    if (nn < 100) v = *(const bf16x8*)(qkv + (size_t)nn * 768 + kb);
    *(bf16x8*)&Qsm[n][kb] = v;
  }
  __syncthreads();

  // ---- logits for head g: 7 m-frags, K=256 in 8 chunks (one per head h) ----
  f32x8 acc[7];
  for (int f = 0; f < 7; ++f) acc[f] = zero8f();
  for (int h = 0; h < 8; ++h) {
    bf16x16 a = ld_frag_a(&Qsm[col][h * 32], half);
    float fac = scale * Wl[g * 8 + h];
    for (int e = 0; e < 16; ++e) a[e] = (__bf16)((float)a[e] * fac);
    for (int f = 0; f < 7; ++f) {
      bf16x16 b = ld_frag_b(&Ksm[f * 16 + col][h * 32], half);
      acc[f] = wmma_bf16f32(a, b, acc[f]);
    }
  }
  // mask m >= 100 (frag 6 holds m = 96 + col)
  if (col >= 4)
    for (int r = 0; r < 8; ++r) acc[6][r] = -1e30f;

  // ---- softmax over m, rows split across lane halves ----
  float mx[8], sm[8];
  for (int r = 0; r < 8; ++r) {
    float m = acc[0][r];
    for (int f = 1; f < 7; ++f) m = fmaxf(m, acc[f][r]);
    mx[r] = m;
  }
  for (int off = 1; off < 16; off <<= 1)
    for (int r = 0; r < 8; ++r) mx[r] = fmaxf(mx[r], __shfl_xor(mx[r], off, 32));
  for (int r = 0; r < 8; ++r) sm[r] = 0.f;
  for (int f = 0; f < 7; ++f)
    for (int r = 0; r < 8; ++r) {
      float e = __expf(acc[f][r] - mx[r]);
      acc[f][r] = e; sm[r] += e;
    }
  for (int off = 1; off < 16; off <<= 1)
    for (int r = 0; r < 8; ++r) sm[r] += __shfl_xor(sm[r], off, 32);
  for (int r = 0; r < 8; ++r) sm[r] = 1.0f / sm[r];
  for (int f = 0; f < 7; ++f)
    for (int r = 0; r < 8; ++r)
      Pbuf[g][half * 8 + r][f * 16 + col] = acc[f][r] * sm[r];
  __syncthreads();

  // ---- talking-heads mix (Ww, bw) + attnscale -> Abuf (m padded to 128) ----
  for (int idx = t; idx < 8 * 16 * 128; idx += 256) {
    int hh = idx >> 11, n = (idx >> 7) & 15, m = idx & 127;
    float v = 0.0f;
    if (m < 100) {
      float s = bw[hh];
      for (int gg = 0; gg < 8; ++gg) s += Ww[hh * 8 + gg] * Pbuf[gg][n][m];
      float l = lamb[hh];
      v = s * (1.0f + l) - 0.01f * l;  // attn_d + (s - attn_d)*(1+l), d=1/100
    }
    Abuf[hh][n][m] = (__bf16)v;
  }
  __syncthreads();

  // ---- restage V transposed into union (Ksm is dead now) ----
  for (int idx = t; idx < 112 * 32; idx += 256) {
    int m = idx >> 5, cb = (idx & 31) * 8;
    bf16x8 v = zero8b();
    if (m < 100) v = *(const bf16x8*)(qkv + (size_t)m * 768 + 512 + cb);
    for (int e = 0; e < 8; ++e) Vt[cb + e][m] = v[e];
  }
  __syncthreads();

  // ---- out_h[16,32] = A_h[16,128] @ V_h[128,32], wave g = head g ----
  f32x8 oacc[2];
  oacc[0] = zero8f(); oacc[1] = zero8f();
  for (int kc = 0; kc < 4; ++kc) {
    bf16x16 a = ld_frag_a(&Abuf[g][col][kc * 32], half);
    for (int f = 0; f < 2; ++f) {
      bf16x16 b = ld_frag_b(&Vt[g * 32 + f * 16 + col][kc * 32], half);
      oacc[f] = wmma_bf16f32(a, b, oacc[f]);
    }
  }
  for (int f = 0; f < 2; ++f) {
    int cc = g * 32 + f * 16 + col;
    for (int r = 0; r < 8; ++r) {
      int nn = n0 + half * 8 + r;
      if (nn < 100)
        AO[((size_t)Bq * 100 + nn) * 256 + cc] = (__bf16)oacc[f][r];
    }
  }
}

// ---------------------------------------------------------------------------
extern "C" void kernel_launch(void* const* d_in, const int* in_sizes, int n_in,
                              void* d_out, int out_size, void* d_ws,
                              size_t ws_size, hipStream_t stream)
{
  (void)in_sizes; (void)n_in; (void)out_size; (void)ws_size;
  const float* buffer  = (const float*)d_in[0];
  const float* spa_pos = (const float*)d_in[1];
  const float* W_mlp   = (const float*)d_in[2];
  const float* g1      = (const float*)d_in[3];
  const float* b1      = (const float*)d_in[4];
  const float* Wqkv    = (const float*)d_in[5];
  const float* Wproj   = (const float*)d_in[6];
  const float* bproj   = (const float*)d_in[7];
  const float* Wl      = (const float*)d_in[8];
  /* d_in[9] = bl: softmax shift-invariant, mathematically a no-op */
  const float* Ww      = (const float*)d_in[10];
  const float* bw      = (const float*)d_in[11];
  const float* lamb    = (const float*)d_in[12];
  const float* g2      = (const float*)d_in[13];
  const float* b2      = (const float*)d_in[14];
  const float* W1      = (const float*)d_in[15];
  const float* W2      = (const float*)d_in[16];
  const float* Wconv   = (const float*)d_in[17];

  char* ws = (char*)d_ws;
  float*  T0  = (float*)(ws);                          // [R,256] f32
  float*  Y1  = (float*)(ws + 104857600);              // [R,256] f32 (also TS)
  __bf16* X1  = (__bf16*)(ws + 209715200);             // [R,256] bf16 (AO/Y2b)
  __bf16* QKV = (__bf16*)(ws + 262144000);             // [R,768] bf16
  __bf16* X2  = QKV;                                   // [R,256] bf16 (reuse)
  __bf16* G   = (__bf16*)(ws + 262144000 + 52428800);  // [R,512] bf16 (reuse)
  float*  TS  = Y1;
  __bf16* AO  = X1;
  __bf16* Y2b = X1;

  dim3 blk(256);
  // 1) T0 = sai2token(buffer)            im2col GEMM [R,1152]x[1152,256]
  gemm_bf16_kernel<<<dim3(1600, 2), blk, 0, stream>>>(
      nullptr, buffer, nullptr, 1, 0, W_mlp, 256, 0, 256, 1152,
      nullptr, nullptr, 0, T0, nullptr, 0);
  // 2) TS = sai2token(buffer + pos)      (linearity of unfold+matmul)
  gemm_bf16_kernel<<<dim3(1600, 2), blk, 0, stream>>>(
      nullptr, buffer, spa_pos, 2, 0, W_mlp, 256, 0, 256, 1152,
      nullptr, nullptr, 0, TS, nullptr, 0);
  // 3) X1 = LN(TS; g1,b1)
  ln_kernel<<<dim3(12800), blk, 0, stream>>>(TS, g1, b1, X1);
  // 4) QKV = X1 @ Wqkv
  gemm_bf16_kernel<<<dim3(1600, 6), blk, 0, stream>>>(
      X1, nullptr, nullptr, 0, 256, Wqkv, 768, 0, 768, 256,
      nullptr, nullptr, 0, nullptr, QKV, 0);
  // 5) attention -> AO
  attn_kernel<<<dim3(1024, 7), blk, 0, stream>>>(QKV, Wl, Ww, bw, lamb, AO);
  // 6) Y1 = AO @ Wproj + bproj + T0      (attention out + residual)
  gemm_bf16_kernel<<<dim3(1600, 2), blk, 0, stream>>>(
      AO, nullptr, nullptr, 0, 256, Wproj, 256, 0, 256, 256,
      bproj, T0, 0, Y1, nullptr, 0);
  // 7) X2 = LN(Y1; g2,b2)
  ln_kernel<<<dim3(12800), blk, 0, stream>>>(Y1, g2, b2, X2);
  // 8) G = relu(X2 @ W1)
  gemm_bf16_kernel<<<dim3(1600, 4), blk, 0, stream>>>(
      X2, nullptr, nullptr, 0, 256, W1, 512, 0, 512, 256,
      nullptr, nullptr, 1, nullptr, G, 0);
  // 9) Y2b = G @ W2 + Y1
  gemm_bf16_kernel<<<dim3(1600, 2), blk, 0, stream>>>(
      G, nullptr, nullptr, 0, 512, W2, 256, 0, 256, 512,
      nullptr, Y1, 0, nullptr, Y2b, 0);
  // 10) out[b,128,25,32,32] = scatter(Y2b @ Wconv^T)   (Wconv is [N=128,K=256])
  gemm_bf16_kernel<<<dim3(1600, 1), blk, 0, stream>>>(
      Y2b, nullptr, nullptr, 0, 256, Wconv, 256, 1, 128, 256,
      nullptr, nullptr, 0, (float*)d_out, nullptr, 1);
}
